// MultiHeadAttention_23476291240050
// MI455X (gfx1250) — matrix operand order repaired and checked by reading, compiled, not run on hardware
//
#include <hip/hip_runtime.h>
#include <cmath>

#define B_  2
#define S_  2048
#define D_  1024
#define H_  16
#define DK_ 64
#define DV_ 64

typedef __attribute__((ext_vector_type(16))) __bf16 v16bf;
typedef __attribute__((ext_vector_type(8)))  float  v8f;
typedef __attribute__((ext_vector_type(2)))  __bf16 bf16x2;

static __device__ __forceinline__ __bf16 f2bf(float x) { return (__bf16)x; }

// ---------------------------------------------------------------------------
// Tiled GEMM: C[4096,1024] = A[4096,1024] @ W[1024,1024], bf16 WMMA, f32 acc.
// EPI==0: write bf16 head-major [B,H,S,64] (QKV projection), scaled.
// EPI==1: write f32 row-major [4096,1024] (final projection).
// Block = 256 threads (8 wave32 waves), tile 128x128, BK=32.
// ---------------------------------------------------------------------------
template <typename AT, int EPI>
__global__ __launch_bounds__(256) void gemm_bf16_kernel(
    const AT* __restrict__ A, const float* __restrict__ W,
    void* __restrict__ outp, float scale)
{
  constexpr int BM = 128, BN = 128, BK = 32, LDT = BK + 2;
  __shared__ __bf16 As[BM * LDT];   // [m][k]
  __shared__ __bf16 Bs[BN * LDT];   // [n][k]  (W tile stored transposed)

  const int tid  = threadIdx.x;
  const int wave = tid >> 5, lane = tid & 31;
  const int wm = wave & 1, wn = wave >> 1;     // 2x4 wave grid over tile
  const int g = lane >> 4, lm = lane & 15;
  const int mBase = blockIdx.x * BM, nBase = blockIdx.y * BN;

  v8f acc[4][2] = {};

  const int ar = tid >> 1, ak = (tid & 1) * 16;      // A fill: 128 rows x 32 k
  const int bk = tid >> 3, bn = (tid & 7) * 16;      // W fill: 32 k x 128 n

  const AT*    apBase = A + (size_t)(mBase + ar) * D_ + ak;
  const float* wpBase = W + (size_t)bk * 1024 + nBase + bn;

  for (int k0 = 0; k0 < D_; k0 += BK) {
    const AT* ap = apBase + k0;
    __bf16* as = &As[ar * LDT + ak];
#pragma unroll
    for (int i = 0; i < 16; ++i) as[i] = f2bf((float)ap[i]);

    const float* wp = wpBase + (size_t)k0 * 1024;
#pragma unroll
    for (int i = 0; i < 16; ++i) Bs[(bn + i) * LDT + bk] = f2bf(wp[i]);

    if (k0 + BK < D_) {             // prefetch next tiles (global_prefetch_b8)
      __builtin_prefetch(ap + BK, 0, 1);
      __builtin_prefetch(wp + (size_t)BK * 1024, 0, 1);
    }
    __syncthreads();

    v16bf afr[4], bfr[2];
#pragma unroll
    for (int mt = 0; mt < 4; ++mt) {
      const __bf16* ab = &As[(wm * 64 + mt * 16 + lm) * LDT];
#pragma unroll
      for (int j = 0; j < 8; ++j) {
        int kk = ((j & 4) ? 16 : 0) + 8 * g + 2 * (j & 3);
        bf16x2 p = *(const bf16x2*)(ab + kk);
        afr[mt][2 * j] = p.x; afr[mt][2 * j + 1] = p.y;
      }
    }
#pragma unroll
    for (int nt = 0; nt < 2; ++nt) {
      const __bf16* bb = &Bs[(wn * 32 + nt * 16 + lm) * LDT];
#pragma unroll
      for (int j = 0; j < 8; ++j) {
        bf16x2 p = *(const bf16x2*)(bb + ((j & 4) ? 16 : 0) + 8 * g + 2 * (j & 3));
        bfr[nt][2 * j] = p.x; bfr[nt][2 * j + 1] = p.y;
      }
    }
#pragma unroll
    for (int mt = 0; mt < 4; ++mt)
#pragma unroll
      for (int nt = 0; nt < 2; ++nt)
        acc[mt][nt] = __builtin_amdgcn_wmma_f32_16x16x32_bf16(
            false, afr[mt], false, bfr[nt], (short)0, acc[mt][nt], false, false);
    __syncthreads();
  }

#pragma unroll
  for (int mt = 0; mt < 4; ++mt)
#pragma unroll
    for (int nt = 0; nt < 2; ++nt)
#pragma unroll
      for (int r = 0; r < 8; ++r) {
        int row = mBase + wm * 64 + mt * 16 + 8 * g + r;
        int col = nBase + wn * 32 + nt * 16 + lm;
        float v = acc[mt][nt][r] * scale;
        if constexpr (EPI == 0) {
          __bf16* out = (__bf16*)outp;
          int bb = row >> 11, ss = row & (S_ - 1);
          int hh = col >> 6,  dd = col & 63;
          out[(((size_t)(bb * H_ + hh) * S_) + ss) * DK_ + dd] = f2bf(v);
        } else {
          float* out = (float*)outp;
          out[(size_t)row * 1024 + col] = v;
        }
      }
}

// ---------------------------------------------------------------------------
// Fused attention: one WG = (b, h, 32 query rows). 8 waves.
// Phase 1: S = Q@K^T (bf16 WMMA), mask (branchless), write RAW scores to attn,
//          keep online per-row (max, sumexp); merge across lanes/waves.
// Phase 2: re-read raw scores, p = exp(s-m)/l, overwrite attn with normalized
//          f32 and accumulate P@V with bf16 WMMA (V staged transposed in LDS).
// ---------------------------------------------------------------------------
__global__ __launch_bounds__(256) void attn_fused_kernel(
    const __bf16* __restrict__ Qh, const __bf16* __restrict__ Kh,
    const __bf16* __restrict__ Vh, const int* __restrict__ mask,
    float* __restrict__ attn, __bf16* __restrict__ ctx)
{
  __shared__ __bf16 Qs[32][DK_ + 2];
  __shared__ int mq[32];
  __shared__ unsigned char mk[S_];
  __shared__ float partM[4][32], partL[4][32];
  __shared__ float rowM[32], rowL[32];
  __shared__ __bf16 Vs[64][34];            // V chunk transposed: [dv][k]

  const int tid  = threadIdx.x;
  const int wave = tid >> 5, lane = tid & 31;
  const int g = lane >> 4, lm = lane & 15;
  const int qbase = blockIdx.x * 32;
  const int h = blockIdx.y, b = blockIdx.z;
  const int bh = b * H_ + h;

  const __bf16* Qp = Qh + (size_t)bh * S_ * DK_;
  const __bf16* Kp = Kh + (size_t)bh * S_ * DK_;
  const __bf16* Vp = Vh + (size_t)bh * S_ * DV_;
  float* attn_bh = attn + (size_t)bh * S_ * S_;

  { // stage Q tile + masks
    int r = tid >> 3, s8 = (tid & 7) * 8;
    const __bf16* src = Qp + (size_t)(qbase + r) * DK_ + s8;
#pragma unroll
    for (int i = 0; i < 8; ++i) Qs[r][s8 + i] = src[i];
  }
  if (tid < 32) mq[tid] = (mask[(size_t)b * S_ + qbase + tid] != 0) ? 1 : 0;
  for (int i = tid; i < S_; i += 256)
    mk[i] = (unsigned char)(mask[(size_t)b * S_ + i] != 0);
  __syncthreads();

  const int mt = wave & 1;        // 16-row group of the 32-row tile
  const int slot = wave >> 1;     // 0..3

  // --- Phase 1 ---
  v16bf qa[2];
#pragma unroll
  for (int kc = 0; kc < 2; ++kc)
#pragma unroll
    for (int j = 0; j < 8; ++j) {
      int kk = 32 * kc + 16 * g + 2 * j;
      bf16x2 p = *(const bf16x2*)(&Qs[16 * mt + lm][kk]);
      qa[kc][2 * j] = p.x; qa[kc][2 * j + 1] = p.y;
    }

  // hoist loop-invariant row masks into registers (no LDS access in hot loop)
  int qm[8];
#pragma unroll
  for (int r = 0; r < 8; ++r) qm[r] = mq[16 * mt + 8 * g + r];

  float pm[8], pl[8];
#pragma unroll
  for (int r = 0; r < 8; ++r) { pm[r] = -3.0e38f; pl[r] = 0.0f; }

  // base pointer for raw-score stores: rows differ by r -> immediate offsets
  float* srow0 = attn_bh + (size_t)(qbase + 16 * mt + 8 * g) * S_;

  for (int t = 0; t < 32; ++t) {
    const int kb = (slot + 4 * t) * 16;        // key tile base
    v8f c = {};
    const __bf16* krow = Kp + (size_t)(kb + lm) * DK_;
#pragma unroll
    for (int kc = 0; kc < 2; ++kc) {
      v16bf bfrag = *(const v16bf*)(krow + 32 * kc + 16 * g);
      c = __builtin_amdgcn_wmma_f32_16x16x32_bf16(
          false, qa[kc], false, bfrag, (short)0, c, false, false);
    }
    const int col = kb + lm;
    const int kcm = (int)mk[col];
    float* sp = srow0 + col;
#pragma unroll
    for (int r = 0; r < 8; ++r) {
      float val = (kcm & qm[r]) ? c[r] : -1.0e30f;   // branchless cndmask
      sp[(size_t)r * S_] = val;                      // RAW score
      float nm = fmaxf(pm[r], val);
      pl[r] = pl[r] * __expf(pm[r] - nm) + __expf(val - nm);
      pm[r] = nm;
    }
  }
  // reduce (m,l) across the 16 lanes holding one row's columns
#pragma unroll
  for (int off = 1; off <= 8; off <<= 1) {
#pragma unroll
    for (int r = 0; r < 8; ++r) {
      float om = __shfl_xor(pm[r], off, 32);
      float ol = __shfl_xor(pl[r], off, 32);
      float nm = fmaxf(pm[r], om);
      pl[r] = pl[r] * __expf(pm[r] - nm) + ol * __expf(om - nm);
      pm[r] = nm;
    }
  }
  if (lm == 0) {
#pragma unroll
    for (int r = 0; r < 8; ++r) {
      int rl = 16 * mt + 8 * g + r;
      partM[slot][rl] = pm[r];
      partL[slot][rl] = pl[r];
    }
  }
  __threadfence();
  __syncthreads();
  if (tid < 32) {
    float m = -3.0e38f, l = 0.0f;
#pragma unroll
    for (int s = 0; s < 4; ++s) {
      float sm = partM[s][tid], sl = partL[s][tid];
      float nm = fmaxf(m, sm);
      l = l * __expf(m - nm) + sl * __expf(sm - nm);
      m = nm;
    }
    rowM[tid] = m; rowL[tid] = l;
  }
  __syncthreads();

  // --- Phase 2 ---
  const int vt = slot;                           // dv tile 0..3
  const float mrow = rowM[16 * mt + lm];
  const float linv = 1.0f / rowL[16 * mt + lm];
  float* arow = attn_bh + (size_t)(qbase + 16 * mt + lm) * S_;
  v8f co = {};

  for (int c = 0; c < 64; ++c) {                 // 32-key chunks
    { // stage V chunk transposed
      int n = tid & 63, ks = (tid >> 6) * 8;
#pragma unroll
      for (int i = 0; i < 8; ++i)
        Vs[n][ks + i] = Vp[(size_t)(32 * c + ks + i) * DV_ + n];
    }
    __syncthreads();

    v16bf pa, pb;
    float2 pv[8];
#pragma unroll
    for (int j = 0; j < 8; ++j) {
      int kk = 32 * c + ((j & 4) ? 16 : 0) + 8 * g + 2 * (j & 3);
      float2 sv = *(const float2*)(arow + kk);
      pv[j].x = __expf(sv.x - mrow) * linv;
      pv[j].y = __expf(sv.y - mrow) * linv;
      pa[2 * j]     = f2bf(pv[j].x);
      pa[2 * j + 1] = f2bf(pv[j].y);
    }
#pragma unroll
    for (int j = 0; j < 8; ++j) {
      bf16x2 p = *(const bf16x2*)(&Vs[16 * vt + lm][((j & 4) ? 16 : 0) + 8 * g + 2 * (j & 3)]);
      pb[2 * j] = p.x; pb[2 * j + 1] = p.y;
    }
    __syncthreads();                 // all raw reads + Vs reads done
    if (vt == 0) {                   // overwrite attn with normalized probs
#pragma unroll
      for (int j = 0; j < 8; ++j) {
        int kk = 32 * c + ((j & 4) ? 16 : 0) + 8 * g + 2 * (j & 3);
        *(float2*)(arow + kk) = pv[j];
      }
    }
    co = __builtin_amdgcn_wmma_f32_16x16x32_bf16(
        false, pa, false, pb, (short)0, co, false, false);
  }

#pragma unroll
  for (int r = 0; r < 8; ++r) {
    int rl = 16 * mt + 8 * g + r;
    int dv = 16 * vt + lm;
    ctx[((size_t)b * S_ + qbase + rl) * (H_ * DV_) + h * DV_ + dv] = f2bf(co[r]);
  }
}

// ---------------------------------------------------------------------------
extern "C" void kernel_launch(void* const* d_in, const int* in_sizes, int n_in,
                              void* d_out, int out_size, void* d_ws, size_t ws_size,
                              hipStream_t stream)
{
  (void)in_sizes; (void)n_in; (void)out_size; (void)ws_size;
  const float* q   = (const float*)d_in[0];
  const float* k   = (const float*)d_in[1];
  const float* v   = (const float*)d_in[2];
  const int*   msk = (const int*)d_in[3];
  const float* Wq  = (const float*)d_in[4];
  const float* Wk  = (const float*)d_in[5];
  const float* Wv  = (const float*)d_in[6];
  const float* Wfc = (const float*)d_in[7];

  float* out_f = (float*)d_out;                       // [B,S,D]
  float* attn  = out_f + (size_t)B_ * S_ * D_;        // [B,H,S,S]

  char* ws = (char*)d_ws;
  const size_t headBytes = (size_t)B_ * H_ * S_ * DK_ * sizeof(__bf16);
  __bf16* Qh  = (__bf16*)(ws);
  __bf16* Kh  = (__bf16*)(ws + headBytes);
  __bf16* Vh  = (__bf16*)(ws + 2 * headBytes);
  __bf16* ctx = (__bf16*)(ws + 3 * headBytes);

  const float scale =
      (float)((log((double)S_) / log(512.0)) / sqrt((double)DK_));

  dim3 gg(32, 8), gb(256);
  gemm_bf16_kernel<float, 0><<<gg, gb, 0, stream>>>(q, Wq, (void*)Qh, scale);
  gemm_bf16_kernel<float, 0><<<gg, gb, 0, stream>>>(k, Wk, (void*)Kh, 1.0f);
  gemm_bf16_kernel<float, 0><<<gg, gb, 0, stream>>>(v, Wv, (void*)Vh, 1.0f);

  dim3 ag(S_ / 32, H_, B_);
  attn_fused_kernel<<<ag, gb, 0, stream>>>(Qh, Kh, Vh, msk, attn, ctx);

  gemm_bf16_kernel<__bf16, 1><<<gg, gb, 0, stream>>>(ctx, Wfc, (void*)out_f, 1.0f);
}
